// LRP_PURE_66108136620350
// MI455X (gfx1250) — matrix operand order repaired and checked by reading, not compiled, not run on hardware
//
#include <hip/hip_runtime.h>

#define HID 64
#define LL 16
#define LAYERS 4
#define NN 50000
#define EE 200000
#define PR 1600000
#define DD (PR / LL)          // 100000
#define KDIM 1024             // L*HID
#define NTILES (DD / 16)      // 6250

typedef __attribute__((ext_vector_type(16))) __bf16 v16bf;
typedef __attribute__((ext_vector_type(8)))  float  v8f;

typedef int v4i __attribute__((vector_size(16)));
typedef __attribute__((address_space(1))) v4i* as1_v4i;
typedef __attribute__((address_space(3))) v4i* as3_v4i;

union Frag { v16bf v; uint4 q[2]; };

__device__ __forceinline__ unsigned f2bf(float f) {
    unsigned u = __builtin_bit_cast(unsigned, f);
    unsigned r = 0x7FFFu + ((u >> 16) & 1u);   // round-to-nearest-even
    return (u + r) >> 16;
}

__device__ __forceinline__ void atomic_pk_add_bf16(unsigned short* addr, unsigned pk) {
    unsigned long long a = (unsigned long long)addr;
    asm volatile("global_atomic_pk_add_bf16 %0, %1, off" :: "v"(a), "v"(pk) : "memory");
}

__device__ __forceinline__ void wait_asynccnt0() {
#if __has_builtin(__builtin_amdgcn_s_wait_asynccnt)
    __builtin_amdgcn_s_wait_asynccnt(0);
#else
    asm volatile("s_wait_asynccnt 0x0" ::: "memory");
#endif
}

// async copy of one 16B chunk global -> LDS (falls back to sync copy)
__device__ __forceinline__ void copy16_g2l(const uint4* g, uint4* l) {
#if __has_builtin(__builtin_amdgcn_global_load_async_to_lds_b128)
    __builtin_amdgcn_global_load_async_to_lds_b128((as1_v4i)g, (as3_v4i)l, 0, 0);
#else
    *l = *g;
#endif
}

// ---------------- h = atom_emb[node_feat] ----------------
__global__ void k_embed(const int* __restrict__ node_feat,
                        const float* __restrict__ atom_emb,
                        float* __restrict__ h) {
    int t = blockIdx.x * 256 + threadIdx.x;
    if (t >= NN * HID) return;
    int n = t >> 6, c = t & 63;
    h[t] = atom_emb[node_feat[n] * HID + c];
}

// ------------- p[row] += val * h[col]  (bf16 pk atomics, wave per entry) -------------
__global__ void k_spmm_n2p(const int* __restrict__ rows, const int* __restrict__ cols,
                           const float* __restrict__ vals, const float* __restrict__ h,
                           unsigned short* __restrict__ p, int nE) {
    int e = blockIdx.x * 8 + (threadIdx.x >> 5);
    if (e >= nE) return;
    int lane = threadIdx.x & 31;
    int row = rows[e], col = cols[e];
    float val = vals[e];
    const float* hr = h + (size_t)col * HID + lane * 2;
    unsigned pk = f2bf(hr[0] * val) | (f2bf(hr[1] * val) << 16);
    atomic_pk_add_bf16(p + (size_t)row * HID + lane * 2, pk);
}

// ------------- p[row] += val * bond_emb[layer][edge_feat[col]] -------------
__global__ void k_spmm_e2p(const int* __restrict__ rows, const int* __restrict__ cols,
                           const float* __restrict__ vals, const int* __restrict__ edge_feat,
                           const float* __restrict__ bond_emb,
                           unsigned short* __restrict__ p, int nE, int layer) {
    int e = blockIdx.x * 8 + (threadIdx.x >> 5);
    if (e >= nE) return;
    int lane = threadIdx.x & 31;
    int row = rows[e], edge = cols[e];
    float val = vals[e];
    const float* er = bond_emb + ((size_t)(layer * 4 + edge_feat[edge])) * HID + lane * 2;
    unsigned pk = f2bf(er[0] * val) | (f2bf(er[1] * val) << 16);
    atomic_pk_add_bf16(p + (size_t)row * HID + lane * 2, pk);
}

// ------------- pack weights[layer] into per-lane WMMA B-fragment layout -------------
// B[k][c] = w[b][c][a], k = a*64+b.  Fragment (nt, kb): lane l holds, in dword j,
// bf16 pair (K = kb*32 + (l/16)*16 + 2j, +1) for column c = nt*16 + l%16.
__global__ void k_pack_b(const float* __restrict__ weights, unsigned* __restrict__ bpack,
                         int layer) {
    int t = blockIdx.x * 256 + threadIdx.x;
    if (t >= 4 * 32 * 32 * 8) return;
    int j    = t & 7;
    int lane = (t >> 3) & 31;
    int kb   = (t >> 8) & 31;
    int nt   = t >> 13;
    int g    = lane >> 4;
    int c    = nt * 16 + (lane & 15);
    int k0   = kb * 32 + g * 16 + 2 * j;
    int k1   = k0 + 1;
    const float* W = weights + (size_t)layer * HID * HID * LL;
    int a0 = k0 >> 6, b0 = k0 & 63;
    int a1 = k1 >> 6, b1 = k1 & 63;
    float f0 = W[((b0 * HID + c) << 4) + a0];
    float f1 = W[((b1 * HID + c) << 4) + a1];
    bpack[t] = f2bf(f0) | (f2bf(f1) << 16);
}

// ------------- GEMM: pd[D,64] = relu( A[D,1024](bf16) @ B[1024,64] + bias ) -------------
// 8 waves / workgroup, one 16-row D-tile per wave.  B fragments staged into LDS in two
// 64 KB K-phases via async global->LDS loads; WMMA reads B from LDS (ds_load_b128).
__global__ void __launch_bounds__(256) k_gemm(const unsigned short* __restrict__ p,
                                              const uint4* __restrict__ bq,
                                              const float* __restrict__ bias,
                                              float* __restrict__ pd, int layer) {
    __shared__ uint4 sB[4096];                    // 64 KB: 4 nt * 16 kb * 32 lanes * 2
    int tid  = threadIdx.x;
    int tile = blockIdx.x * 8 + (tid >> 5);       // wave-uniform
    int lane = tid & 31;
    int m    = lane & 15;
    int g    = lane >> 4;
    int dbase = tile * 16;
    bool active = (tile < NTILES);

    const unsigned short* Arow = p + (size_t)(dbase + m) * KDIM;

    v8f acc[4];
#pragma unroll
    for (int nt = 0; nt < 4; ++nt)
#pragma unroll
        for (int r = 0; r < 8; ++r) acc[nt][r] = 0.0f;

    for (int ph = 0; ph < 2; ++ph) {
        // stage kb in [ph*16, ph*16+16) for all 4 nt: 4096 uint4, 16 per thread
#pragma unroll
        for (int it = 0; it < 16; ++it) {
            int idx = it * 256 + tid;             // LDS uint4 index
            int q   = idx & 1;
            int ln  = (idx >> 1) & 31;
            int kbl = (idx >> 6) & 15;
            int nt  = idx >> 10;
            int gidx = (((nt * 32 + (ph * 16 + kbl)) * 32) + ln) * 2 + q;
            copy16_g2l(bq + gidx, &sB[idx]);
        }
        wait_asynccnt0();
        __syncthreads();

        if (active) {
#pragma unroll 2
            for (int kbl = 0; kbl < 16; ++kbl) {
                int kb = ph * 16 + kbl;
                Frag af;
                af.q[0] = *(const uint4*)(Arow + kb * 32 + g * 16);
                af.q[1] = *(const uint4*)(Arow + kb * 32 + g * 16 + 8);
#pragma unroll
                for (int nt = 0; nt < 4; ++nt) {
                    Frag bf;
                    const uint4* bp = &sB[(((nt * 16 + kbl) * 32) + lane) * 2];
                    bf.q[0] = bp[0];
                    bf.q[1] = bp[1];
                    acc[nt] = __builtin_amdgcn_wmma_f32_16x16x32_bf16(
                        false, af.v, false, bf.v, (short)0, acc[nt], false, false);
                }
            }
        }
        __syncthreads();
    }

    if (!active) return;
    const float* bl = bias + layer * HID;
#pragma unroll
    for (int nt = 0; nt < 4; ++nt) {
        int c = nt * 16 + (lane & 15);
        float bv = bl[c];
#pragma unroll
        for (int r = 0; r < 8; ++r) {
            int row = dbase + r + 8 * g;
            pd[(size_t)row * HID + c] = fmaxf(acc[nt][r] + bv, 0.0f);
        }
    }
}

// ------------- pooled[row] += val * pd[col]  (f32 atomics, wave per entry) -------------
__global__ void k_pool(const int* __restrict__ rows, const int* __restrict__ cols,
                       const float* __restrict__ vals, const float* __restrict__ pd,
                       float* __restrict__ pooled, int nJ) {
    int j = blockIdx.x * 8 + (threadIdx.x >> 5);
    if (j >= nJ) return;
    int lane = threadIdx.x & 31;
    int row = rows[j], col = cols[j];
    float val = vals[j];
    const float* pr = pd + (size_t)col * HID + lane * 2;
    float* o = pooled + (size_t)row * HID + lane * 2;
    atomicAdd(o, val * pr[0]);
    atomicAdd(o + 1, val * pr[1]);
}

// ------------- degree gate: h = pooled * (relu(deg*w0+b0) @ w1^T + b1) -------------
__global__ void k_gate(const float* __restrict__ pooled, const float* __restrict__ degs,
                       const float* __restrict__ w0, const float* __restrict__ b0,
                       const float* __restrict__ w1, const float* __restrict__ b1,
                       float* __restrict__ h, int layer) {
    __shared__ float s_w0[128], s_b0[128], s_w1t[128 * 64];
    int tid = threadIdx.x;
    const float* W0 = w0 + layer * 128;
    const float* B0 = b0 + layer * 128;
    const float* W1 = w1 + (size_t)layer * HID * 128;
    if (tid < 128) { s_w0[tid] = W0[tid]; s_b0[tid] = B0[tid]; }
    for (int idx = tid; idx < 128 * 64; idx += 256) {
        int k = idx >> 6, c = idx & 63;
        s_w1t[idx] = W1[c * 128 + k];     // transposed: lanes hit consecutive banks
    }
    __syncthreads();
    int t = blockIdx.x * 256 + tid;
    if (t >= NN * HID) return;
    int n = t >> 6, c = t & 63;
    float deg = degs[n];
    float sum = b1[layer * HID + c];
#pragma unroll 8
    for (int k = 0; k < 128; ++k) {
        float f = fmaxf(fmaf(deg, s_w0[k], s_b0[k]), 0.0f);
        sum = fmaf(f, s_w1t[k * 64 + c], sum);
    }
    h[t] = pooled[t] * sum;
}

// ------------- column sums of h -------------
__global__ void k_reduce(const float* __restrict__ h, float* __restrict__ accum) {
    __shared__ float s[256];
    int tid = threadIdx.x;
    int c = tid & 63;
    float sum = 0.0f;
    for (int n = blockIdx.x * 4 + (tid >> 6); n < NN; n += gridDim.x * 4)
        sum += h[(size_t)n * HID + c];
    s[tid] = sum;
    __syncthreads();
    if (tid < 64) {
        float tot = s[tid] + s[tid + 64] + s[tid + 128] + s[tid + 192];
        atomicAdd(&accum[tid], tot);
    }
}

__global__ void k_final(const float* __restrict__ accum, const float* __restrict__ fw,
                        const float* __restrict__ fb, float* __restrict__ out) {
    __shared__ float s[64];
    int t = threadIdx.x;
    s[t] = accum[t] * (1.0f / (float)NN) * fw[t];
    __syncthreads();
    for (int off = 32; off > 0; off >>= 1) {
        if (t < off) s[t] += s[t + off];
        __syncthreads();
    }
    if (t == 0) out[0] = s[0] + fb[0];
}

extern "C" void kernel_launch(void* const* d_in, const int* in_sizes, int n_in,
                              void* d_out, int out_size, void* d_ws, size_t ws_size,
                              hipStream_t stream) {
    (void)in_sizes; (void)n_in; (void)out_size; (void)ws_size;
    const int*   node_feat = (const int*)d_in[0];
    const int*   edge_feat = (const int*)d_in[1];
    const float* degs      = (const float*)d_in[2];
    const int*   n2p_rows  = (const int*)d_in[3];
    const int*   n2p_cols  = (const int*)d_in[4];
    const float* n2p_vals  = (const float*)d_in[5];
    const int*   e2p_rows  = (const int*)d_in[6];
    const int*   e2p_cols  = (const int*)d_in[7];
    const float* e2p_vals  = (const float*)d_in[8];
    const int*   pool_rows = (const int*)d_in[9];
    const int*   pool_cols = (const int*)d_in[10];
    const float* pool_vals = (const float*)d_in[11];
    const float* atom_emb  = (const float*)d_in[13];
    const float* bond_emb  = (const float*)d_in[14];
    const float* weights   = (const float*)d_in[15];
    const float* bias      = (const float*)d_in[16];
    const float* deg0_w    = (const float*)d_in[17];
    const float* deg0_b    = (const float*)d_in[18];
    const float* deg1_w    = (const float*)d_in[19];
    const float* deg1_b    = (const float*)d_in[20];
    const float* final_w   = (const float*)d_in[21];
    const float* final_b   = (const float*)d_in[22];

    // workspace layout (all chunks 256B-aligned)
    char* ws = (char*)d_ws;
    size_t off = 0;
    auto take = [&](size_t bytes) { char* r = ws + off; off += (bytes + 255) & ~(size_t)255; return r; };
    float*          h      = (float*)take((size_t)NN * HID * 4);          // 12.8 MB
    unsigned short* p      = (unsigned short*)take((size_t)PR * HID * 2); // 204.8 MB
    float*          pd     = (float*)take((size_t)DD * HID * 4);          // 25.6 MB
    float*          pooled = (float*)take((size_t)NN * HID * 4);          // 12.8 MB
    unsigned*       bpack  = (unsigned*)take(4 * 32 * 32 * 8 * 4);        // 128 KB
    float*          accum  = (float*)take(64 * 4);

    k_embed<<<(NN * HID + 255) / 256, 256, 0, stream>>>(node_feat, atom_emb, h);

    const int PR2 = PR / 2;
    for (int layer = 0; layer < LAYERS; ++layer) {
        (void)hipMemsetAsync(p, 0, (size_t)PR * HID * 2, stream);
        k_spmm_n2p<<<(PR + 7) / 8, 256, 0, stream>>>(n2p_rows, n2p_cols, n2p_vals, h, p, PR);
        k_spmm_e2p<<<(PR2 + 7) / 8, 256, 0, stream>>>(e2p_rows, e2p_cols, e2p_vals,
                                                      edge_feat, bond_emb, p, PR2, layer);
        k_pack_b<<<(4 * 32 * 32 * 8 + 255) / 256, 256, 0, stream>>>(weights, bpack, layer);
        k_gemm<<<(NTILES + 7) / 8, 256, 0, stream>>>(p, (const uint4*)bpack, bias, pd, layer);
        (void)hipMemsetAsync(pooled, 0, (size_t)NN * HID * 4, stream);
        k_pool<<<(DD + 7) / 8, 256, 0, stream>>>(pool_rows, pool_cols, pool_vals, pd, pooled, DD);
        k_gate<<<(NN * HID + 255) / 256, 256, 0, stream>>>(pooled, degs, deg0_w, deg0_b,
                                                           deg1_w, deg1_b, h, layer);
    }

    (void)hipMemsetAsync(accum, 0, 64 * 4, stream);
    k_reduce<<<128, 256, 0, stream>>>(h, accum);
    k_final<<<1, 64, 0, stream>>>(accum, final_w, final_b, (float*)d_out);
}
